// AttentionBlock_44513041055998
// MI455X (gfx1250) — compile-verified, matrix operand order checked by reading, not run on hardware
//
#include <hip/hip_runtime.h>

typedef __attribute__((ext_vector_type(16))) __bf16          v16bf;
typedef __attribute__((ext_vector_type(8)))  float           v8f;
typedef __attribute__((ext_vector_type(16))) unsigned short  v16us;
typedef __attribute__((ext_vector_type(8)))  unsigned short  v8us;

#define B_   8
#define C_   256
#define CR_  8
#define H_   128
#define W_   128
#define HD_  64
#define WD_  64
#define N_   4096   // HD_*WD_

// round-to-nearest-even f32 -> bf16 bits
__device__ __forceinline__ unsigned short f2bf(float f) {
  unsigned int u = __builtin_bit_cast(unsigned int, f);
  u += 0x7FFFu + ((u >> 16) & 1u);
  return (unsigned short)(u >> 16);
}
__device__ __forceinline__ v16bf bfc(v16us u) { return __builtin_bit_cast(v16bf, u); }
__device__ __forceinline__ v16us cat8(v8us a, v8us b) {
  return __builtin_shufflevector(a, b, 0, 1, 2, 3, 4, 5, 6, 7,
                                       8, 9, 10, 11, 12, 13, 14, 15);
}

// ---------------------------------------------------------------------------
// K1: 2x bilinear downsample (half-pixel, scale 2 => exact 2x2 average pool)
// x[B,C,128,128] -> xd[B,C,64,64] (f32, flat [b][c][n])
// ---------------------------------------------------------------------------
__global__ void down_kernel(const float* __restrict__ x, float* __restrict__ xd) {
  int gid = blockIdx.x * 256 + threadIdx.x;               // B*C*64*64 = 8388608
  int xo = gid & (WD_ - 1);
  int r  = gid >> 6;
  int yo = r & (HD_ - 1);
  int bc = r >> 6;
  const float* p = x + (size_t)bc * (H_ * W_) + (size_t)(yo * 2) * W_ + xo * 2;
  xd[gid] = 0.25f * (p[0] + p[1] + p[W_] + p[W_ + 1]);
}

// ---------------------------------------------------------------------------
// K0: swizzle Wv[256,256] f32 into per-lane bf16 A-operand tiles.
// Tile (mt,ks): lane l holds row m = mt*16 + (l&15); K elems:
//   lanes 0-15 : k = ks*32 + {0..7, 16..23};  lanes 16-31: k = ks*32 + {8..15, 24..31}
// Stored contiguously: Wvsw[(((mt*8+ks)*32 + l)*16 + j]
// ---------------------------------------------------------------------------
__global__ void wvprep_kernel(const float* __restrict__ Wv, unsigned short* __restrict__ Wvsw) {
  int gid = blockIdx.x * 256 + threadIdx.x;               // 65536
  int j    = gid & 15;
  int l    = (gid >> 4) & 31;
  int ks   = (gid >> 9) & 7;
  int mt   = gid >> 12;
  int half = l >> 4;
  int cout = mt * 16 + (l & 15);
  int cin  = ks * 32 + half * 8 + ((j < 8) ? j : (16 + (j - 8)));
  Wvsw[gid] = f2bf(Wv[cout * C_ + cin]);
}

// ---------------------------------------------------------------------------
// K2: q/k 1x1 conv (Cr=8, tiny): one thread per (b,n).
// Outputs Qt[b][n][8], Ktt[b][n][8] bf16 (contiguous 16B rows).
// ---------------------------------------------------------------------------
__global__ __launch_bounds__(128) void qkproj_kernel(
    const float* __restrict__ xd,
    const float* __restrict__ Wq, const float* __restrict__ bq,
    const float* __restrict__ Wk, const float* __restrict__ bk,
    unsigned short* __restrict__ Qt, unsigned short* __restrict__ Ktt) {
  __shared__ float wq[CR_ * C_];
  __shared__ float wk[CR_ * C_];
  int t = threadIdx.x;
  for (int i = t; i < CR_ * C_; i += 128) { wq[i] = Wq[i]; wk[i] = Wk[i]; }
  __syncthreads();

  int gid = blockIdx.x * 128 + t;                         // B*N = 32768
  int b = gid >> 12;
  int n = gid & (N_ - 1);
  float qa[CR_], ka[CR_];
#pragma unroll
  for (int cr = 0; cr < CR_; ++cr) { qa[cr] = bq[cr]; ka[cr] = bk[cr]; }
  const float* xp = xd + (size_t)b * C_ * N_ + n;
  for (int c = 0; c < C_; ++c) {
    float xv = xp[(size_t)c * N_];
#pragma unroll
    for (int cr = 0; cr < CR_; ++cr) {
      qa[cr] = fmaf(wq[cr * C_ + c], xv, qa[cr]);
      ka[cr] = fmaf(wk[cr * C_ + c], xv, ka[cr]);
    }
  }
#pragma unroll
  for (int cr = 0; cr < CR_; ++cr) {
    Qt [(size_t)gid * CR_ + cr] = f2bf(qa[cr]);
    Ktt[(size_t)gid * CR_ + cr] = f2bf(ka[cr]);
  }
}

// ---------------------------------------------------------------------------
// K3: V projection via WMMA bf16: V = Wv * xd + bv, output Vcn[b][c][n] bf16.
// One wave computes 256 channels x 16 pixels (16 M-tiles, K loop 8x32).
// All 16 A tiles for a K-step are preloaded in one clause; the f32 B gather +
// bf16 conversion VALU overlaps the load latency.
// ---------------------------------------------------------------------------
__global__ __launch_bounds__(32)
__attribute__((amdgpu_waves_per_eu(1)))
void vproj_kernel(
    const unsigned short* __restrict__ Wvsw, const float* __restrict__ xd,
    const float* __restrict__ bv, unsigned short* __restrict__ Vcn) {
  int wg = blockIdx.x;                                    // B * 256
  int b  = wg >> 8;
  int nt = wg & 255;
  int n0 = nt * 16;
  int l = threadIdx.x, half = l >> 4, lane16 = l & 15;

  __shared__ float bvs[C_];
  for (int i = l; i < C_; i += 32) bvs[i] = bv[i];
  __syncthreads();

  v8f acc[16];
#pragma unroll
  for (int mt = 0; mt < 16; ++mt) acc[mt] = (v8f)0.0f;

  const float* xb = xd + (size_t)b * C_ * N_;
  for (int ks = 0; ks < 8; ++ks) {
    int cin0 = ks * 32 + half * 8;
    const unsigned short* wbase = Wvsw + (size_t)(ks * 32 + l) * 16;

    // issue all 16 A-tile pairs (MLP) ...
    v8us aa0[16], aa1[16];
#pragma unroll
    for (int mt = 0; mt < 16; ++mt) {
      const v8us* np = (const v8us*)(wbase + (size_t)mt * (8 * 32 * 16));
      aa0[mt] = np[0];
      aa1[mt] = np[1];
    }
    // ... and the B gather; convert f32->bf16 while A loads are in flight
    float xf[16];
#pragma unroll
    for (int j = 0; j < 8; ++j) xf[j]     = xb[(size_t)(cin0 + j) * N_ + n0 + lane16];
#pragma unroll
    for (int j = 0; j < 8; ++j) xf[8 + j] = xb[(size_t)(cin0 + 16 + j) * N_ + n0 + lane16];
    v16us bu = (v16us)0;
#pragma unroll
    for (int j = 0; j < 16; ++j) bu[j] = f2bf(xf[j]);
    v16bf Bv = bfc(bu);

#pragma unroll
    for (int mt = 0; mt < 16; ++mt)
      acc[mt] = __builtin_amdgcn_wmma_f32_16x16x32_bf16(
          false, bfc(cat8(aa0[mt], aa1[mt])), false, Bv, (short)0, acc[mt],
          false, false);
  }

#pragma unroll
  for (int mt = 0; mt < 16; ++mt) {
#pragma unroll
    for (int r = 0; r < 8; ++r) {
      int c = mt * 16 + r + 8 * half;
      Vcn[(size_t)(b * C_ + c) * N_ + n0 + lane16] = f2bf(acc[mt][r] + bvs[c]);
    }
  }
}

// ---------------------------------------------------------------------------
// K4: flash attention. One wave per (batch, 16-row query tile).
// Streams 128 key blocks of 32.  Per block: 2 score WMMAs, then ALL 16 V-tile
// pairs are issued immediately (their addresses don't depend on P) so the L2
// latency is hidden behind the ~150 VALU ops of online softmax + LDS
// transpose; finally the 16-WMMA P*V chain consumes them.  O[b][m][c] f32.
// ---------------------------------------------------------------------------
__global__ __launch_bounds__(32)
__attribute__((amdgpu_waves_per_eu(1)))
void flash_kernel(
    const unsigned short* __restrict__ Qt, const unsigned short* __restrict__ Ktt,
    const unsigned short* __restrict__ Vcn, float* __restrict__ O) {
  int wg = blockIdx.x;                                    // B * 256
  int b  = wg >> 8;
  int mt = wg & 255;
  int m0 = mt * 16;
  int l = threadIdx.x, half = l >> 4, lane16 = l & 15;

  __shared__ __attribute__((aligned(16))) unsigned short p_lds[16 * 32];

  // A operand: Q rows m0..m0+15, k = cr (8 real, rest zero); lanes>=16 all zero
  v8us zero8 = (v8us)0;
  v8us q8 = zero8;
  if (half == 0)
    q8 = *(const v8us*)(Qt + (size_t)(b * N_ + m0 + lane16) * CR_);
  v16bf Qa = bfc(cat8(q8, zero8));

  v8f acc[16];
#pragma unroll
  for (int ct = 0; ct < 16; ++ct) acc[ct] = (v8f)0.0f;
  v8f rmax = (v8f)(-3.0e38f);
  v8f rsum = (v8f)0.0f;

  const unsigned short* kbase = Ktt + (size_t)b * N_ * CR_;
  const unsigned short* vbase = Vcn + (size_t)b * C_ * N_ + (size_t)lane16 * N_;
  int koff = half * 8;

  for (int kb = 0; kb < N_ / 32; ++kb) {
    int n0 = kb * 32;

    // ---- scores: S = Q * K^T for 32 keys (two 16-col WMMAs) ----
    v16us ku0 = (v16us)0, ku1 = (v16us)0;
    if (half == 0) {
      v8us k0 = *(const v8us*)(kbase + (size_t)(n0 + lane16) * CR_);
      v8us k1 = *(const v8us*)(kbase + (size_t)(n0 + 16 + lane16) * CR_);
      ku0 = cat8(k0, zero8);
      ku1 = cat8(k1, zero8);
    }
    v8f S0 = __builtin_amdgcn_wmma_f32_16x16x32_bf16(
        false, Qa, false, bfc(ku0), (short)0, (v8f)0.0f, false, false);
    v8f S1 = __builtin_amdgcn_wmma_f32_16x16x32_bf16(
        false, Qa, false, bfc(ku1), (short)0, (v8f)0.0f, false, false);

    // ---- issue ALL V-tile loads now; softmax VALU below hides the latency --
    const unsigned short* vp0 = vbase + n0 + koff;
    v8us vv0[16], vv1[16];
#pragma unroll
    for (int ct = 0; ct < 16; ++ct) {
      const unsigned short* vp = vp0 + (size_t)ct * (16 * N_);
      vv0[ct] = *(const v8us*)vp;
      vv1[ct] = *(const v8us*)(vp + 16);
    }

    // prefetch next key block's K rows into cache
    if (kb + 1 < N_ / 32 && half == 0)
      __builtin_prefetch(kbase + (size_t)(n0 + 32 + lane16) * CR_, 0, 1);

    // ---- online softmax: row r of this lane-half (m = r + 8*half) ----
    v8f t;
#pragma unroll
    for (int r = 0; r < 8; ++r) t[r] = fmaxf(S0[r], S1[r]);
#pragma unroll
    for (int off = 1; off < 16; off <<= 1)
#pragma unroll
      for (int r = 0; r < 8; ++r) t[r] = fmaxf(t[r], __shfl_xor(t[r], off, 32));

    v8f nmax, alpha;
#pragma unroll
    for (int r = 0; r < 8; ++r) {
      nmax[r]  = fmaxf(rmax[r], t[r]);
      alpha[r] = __expf(rmax[r] - nmax[r]);
      rmax[r]  = nmax[r];
    }
    v8f P0, P1, ps;
#pragma unroll
    for (int r = 0; r < 8; ++r) {
      P0[r] = __expf(S0[r] - nmax[r]);
      P1[r] = __expf(S1[r] - nmax[r]);
      ps[r] = P0[r] + P1[r];
    }
#pragma unroll
    for (int off = 1; off < 16; off <<= 1)
#pragma unroll
      for (int r = 0; r < 8; ++r) ps[r] += __shfl_xor(ps[r], off, 32);
#pragma unroll
    for (int r = 0; r < 8; ++r) rsum[r] = rsum[r] * alpha[r] + ps[r];
#pragma unroll
    for (int ct = 0; ct < 16; ++ct) acc[ct] *= alpha;

    // ---- transpose P (C-layout) -> A-layout via LDS ----
#pragma unroll
    for (int r = 0; r < 8; ++r) {
      int m = r + 8 * half;
      p_lds[m * 32 + lane16]      = f2bf(P0[r]);
      p_lds[m * 32 + 16 + lane16] = f2bf(P1[r]);
    }
    __syncthreads();
    {
      int base0 = half * 8;
      v8us pa0 = *(const v8us*)(p_lds + lane16 * 32 + base0);
      v8us pa1 = *(const v8us*)(p_lds + lane16 * 32 + base0 + 16);
      v16bf Pa = bfc(cat8(pa0, pa1));
      __syncthreads();

      // ---- O += P * V over all 256 channels (16 N-tiles) ----
#pragma unroll
      for (int ct = 0; ct < 16; ++ct)
        acc[ct] = __builtin_amdgcn_wmma_f32_16x16x32_bf16(
            false, Pa, false, bfc(cat8(vv0[ct], vv1[ct])), (short)0, acc[ct],
            false, false);
    }
  }

  // ---- epilogue: normalize and store O[b][m][c] ----
  v8f inv;
#pragma unroll
  for (int r = 0; r < 8; ++r) inv[r] = 1.0f / rsum[r];
#pragma unroll
  for (int ct = 0; ct < 16; ++ct)
#pragma unroll
    for (int r = 0; r < 8; ++r) {
      int m = m0 + r + 8 * half;
      int c = ct * 16 + lane16;
      O[((size_t)b * N_ + m) * C_ + c] = acc[ct][r] * inv[r];
    }
}

// ---------------------------------------------------------------------------
// K5: 2x bilinear upsample (half-pixel, clamped) + residual:
// out = x + gamma * up(O);  O is [b][m=y*64+x][c] f32.
// ---------------------------------------------------------------------------
__global__ void up_kernel(const float* __restrict__ x, const float* __restrict__ O,
                          const float* __restrict__ gamma, float* __restrict__ out) {
  size_t gid = (size_t)blockIdx.x * 256 + threadIdx.x;    // B*C*128*128
  int w = (int)(gid & (W_ - 1));
  size_t r = gid >> 7;
  int h = (int)(r & (H_ - 1));
  size_t bc = r >> 7;
  int c = (int)(bc & (C_ - 1));
  int b = (int)(bc >> 8);

  float sy = h * 0.5f - 0.25f;
  float sx = w * 0.5f - 0.25f;
  int y0 = (int)floorf(sy), x0 = (int)floorf(sx);
  float wy = sy - (float)y0, wx = sx - (float)x0;
  int y0c = max(y0, 0), y1c = min(y0 + 1, HD_ - 1);
  int x0c = max(x0, 0), x1c = min(x0 + 1, WD_ - 1);

  const float* Ob = O + (size_t)b * N_ * C_ + c;
  float o00 = Ob[(size_t)(y0c * WD_ + x0c) * C_];
  float o01 = Ob[(size_t)(y0c * WD_ + x1c) * C_];
  float o10 = Ob[(size_t)(y1c * WD_ + x0c) * C_];
  float o11 = Ob[(size_t)(y1c * WD_ + x1c) * C_];
  float up = (1.0f - wy) * ((1.0f - wx) * o00 + wx * o01) +
             wy * ((1.0f - wx) * o10 + wx * o11);
  out[gid] = fmaf(gamma[0], up, x[gid]);
}

// ---------------------------------------------------------------------------
extern "C" void kernel_launch(void* const* d_in, const int* in_sizes, int n_in,
                              void* d_out, int out_size, void* d_ws, size_t ws_size,
                              hipStream_t stream) {
  const float* x     = (const float*)d_in[0];
  const float* Wq    = (const float*)d_in[1];
  const float* bq    = (const float*)d_in[2];
  const float* Wk    = (const float*)d_in[3];
  const float* bk    = (const float*)d_in[4];
  const float* Wv    = (const float*)d_in[5];
  const float* bv    = (const float*)d_in[6];
  const float* gamma = (const float*)d_in[7];
  float* out = (float*)d_out;
  (void)in_sizes; (void)n_in; (void)out_size; (void)ws_size;

  char* ws = (char*)d_ws;
  float*          xd   = (float*)(ws + 0);                    // 33554432 B
  float*          O    = (float*)(ws + 33554432);             // 33554432 B
  unsigned short* Vcn  = (unsigned short*)(ws + 67108864);    // 16777216 B
  unsigned short* Qt   = (unsigned short*)(ws + 83886080);    //   524288 B
  unsigned short* Ktt  = (unsigned short*)(ws + 84410368);    //   524288 B
  unsigned short* Wvsw = (unsigned short*)(ws + 84934656);    //   131072 B

  down_kernel  <<<B_ * C_ * HD_ * WD_ / 256, 256, 0, stream>>>(x, xd);
  wvprep_kernel<<<C_ * C_ / 256,            256, 0, stream>>>(Wv, Wvsw);
  qkproj_kernel<<<B_ * N_ / 128,            128, 0, stream>>>(xd, Wq, bq, Wk, bk, Qt, Ktt);
  vproj_kernel <<<B_ * (N_ / 16),            32, 0, stream>>>(Wvsw, xd, bv, Vcn);
  flash_kernel <<<B_ * (N_ / 16),            32, 0, stream>>>(Qt, Ktt, Vcn, O);
  up_kernel    <<<B_ * C_ * H_ * W_ / 256,  256, 0, stream>>>(x, O, gamma, out);
}
